// E8RHTLinear_3212635537648
// MI455X (gfx1250) — matrix operand order repaired and checked by reading, not compiled, not run on hardware
//
#include <hip/hip_runtime.h>
#include <hip/hip_bf16.h>

typedef __attribute__((ext_vector_type(16))) __bf16 v16bf;
typedef __attribute__((ext_vector_type(8)))  float  v8f;

// CDNA5 async global->LDS path (ASYNCcnt), gated on builtin availability so we
// fall back to the proven sync path if the toolchain lacks the builtins.
#if __has_builtin(__builtin_amdgcn_global_load_async_to_lds_b128) && \
    __has_builtin(__builtin_amdgcn_s_wait_asynccnt)
#define USE_ASYNC_LDS 1
#else
#define USE_ASYNC_LDS 0
#endif

// ---------------------------------------------------------------------------
// Kernel 1: t = FWHT_N(x * SV) * (1/sqrt(N)), output bf16. One block per row.
// N = 4096 (16 KB f32 in LDS).
// ---------------------------------------------------------------------------
__global__ __launch_bounds__(256) void fwht_x_kernel(
    const float* __restrict__ x, const float* __restrict__ SV,
    __bf16* __restrict__ tq, int N)
{
    __shared__ float lds[4096];
    const int row = blockIdx.x;
    const float* xr = x + (size_t)row * N;

    for (int i = threadIdx.x; i < N; i += 256)
        lds[i] = xr[i] * SV[i];
    __syncthreads();

    for (int h = 1; h < N; h <<= 1) {
        for (int t = threadIdx.x; t < (N >> 1); t += 256) {
            const int i = ((t & ~(h - 1)) << 1) | (t & (h - 1));
            const float a = lds[i], b = lds[i + h];
            lds[i]     = a + b;
            lds[i + h] = a - b;
        }
        __syncthreads();
    }

    const float s = rsqrtf((float)N);
    __bf16* out = tq + (size_t)row * N;
    for (int i = threadIdx.x; i < N; i += 256)
        out[i] = (__bf16)(lds[i] * s);
}

// ---------------------------------------------------------------------------
// Kernel 2: W_hat = codebook[Qidxs] + irs * codebook2[Qidxs2], bf16 output.
// One thread per 8-element group; codebooks (2 MB each) stay L2-resident.
// ---------------------------------------------------------------------------
__global__ __launch_bounds__(256) void dequant_kernel(
    const int* __restrict__ q1, const int* __restrict__ q2,
    const float* __restrict__ cb1, const float* __restrict__ cb2,
    const float* __restrict__ irs_p, __bf16* __restrict__ wq,
    long long ngroups)
{
    const long long g = (long long)blockIdx.x * 256 + threadIdx.x;
    if (g >= ngroups) return;
    const float irs = irs_p[0];

    const float4* c1 = (const float4*)(cb1 + (size_t)q1[g] * 8);
    const float4* c2 = (const float4*)(cb2 + (size_t)q2[g] * 8);
    const float4 a0 = c1[0], a1 = c1[1];
    const float4 b0 = c2[0], b1 = c2[1];

    union { __bf16 h[8]; uint4 u; } r;
    r.h[0] = (__bf16)(a0.x + irs * b0.x);
    r.h[1] = (__bf16)(a0.y + irs * b0.y);
    r.h[2] = (__bf16)(a0.z + irs * b0.z);
    r.h[3] = (__bf16)(a0.w + irs * b0.w);
    r.h[4] = (__bf16)(a1.x + irs * b1.x);
    r.h[5] = (__bf16)(a1.y + irs * b1.y);
    r.h[6] = (__bf16)(a1.z + irs * b1.z);
    r.h[7] = (__bf16)(a1.w + irs * b1.w);
    *(uint4*)(wq + (size_t)g * 8) = r.u;
}

// ---------------------------------------------------------------------------
// Kernel 3: z = t @ W^T via v_wmma_f32_16x16x32_bf16.
// t: (T x N) bf16 row-major, W: (Mout x N) bf16 row-major -> both K-contiguous.
// Workgroup tile 128(T) x 128(Mout), K-step 32, 8 waves (4 along T x 2 along M),
// each wave computes 2x4 WMMA 16x16 tiles. Async path: double-buffered LDS fed
// by GLOBAL_LOAD_ASYNC_TO_LDS_B128, pipelined one K-tile ahead on ASYNCcnt.
// ---------------------------------------------------------------------------
#define BM 128
#define BN 128
#define BK 32
#define LDSW 40   // padded row stride in halfs (80 B, 16B-aligned)

union Frag { uint4 u[2]; v16bf v; };

__global__ __launch_bounds__(256) void gemm_kernel(
    const __bf16* __restrict__ tA, const __bf16* __restrict__ tB,
    float* __restrict__ Z, int Mout, int N)
{
    const int tid  = threadIdx.x;
    const int wave = tid >> 5;
    const int lane = tid & 31;
    const int lm   = lane & 15;   // row/col-in-tile lane index
    const int lh   = lane >> 4;   // K-half selector
    const int wm   = wave & 3;    // wave row (T dim): 4 waves * 32 rows = 128
    const int wn   = wave >> 2;   // wave col (M dim): 2 waves * 64 cols = 128

    const int rowBase = blockIdx.y * BM;  // rows of t (T dim)
    const int colBase = blockIdx.x * BN;  // rows of W (Mout dim)

    v8f acc[2][4];
#pragma unroll
    for (int i = 0; i < 2; ++i)
#pragma unroll
        for (int j = 0; j < 4; ++j)
            acc[i][j] = (v8f){0.f, 0.f, 0.f, 0.f, 0.f, 0.f, 0.f, 0.f};

    // Each thread owns one 32B chunk (16 halfs) of A and of B per K-step.
    const int ldRow  = tid >> 1;            // 0..127
    const int ldHalf = (tid & 1) * 16;      // 0 or 16 (halfs)
    const __bf16* gA = tA + (size_t)(rowBase + ldRow) * N + ldHalf;
    const __bf16* gB = tB + (size_t)(colBase + ldRow) * N + ldHalf;

#if USE_ASYNC_LDS
    __shared__ __bf16 sA[2][BM * LDSW];
    __shared__ __bf16 sB[2][BN * LDSW];

    // Builtin signature (from toolchain diagnostics): pointee is int
    // __attribute__((vector_size(16))); src is addrspace(1), dst addrspace(3).
    typedef int v4i_t __attribute__((vector_size(16)));
    typedef __attribute__((address_space(1))) v4i_t* gv4i_p;
    typedef __attribute__((address_space(3))) v4i_t* lv4i_p;

#define ISSUE_TILE(buf, kk)                                                   \
    do {                                                                      \
        gv4i_p ga_ = (gv4i_p)(gA + (kk));                                     \
        gv4i_p gb_ = (gv4i_p)(gB + (kk));                                     \
        lv4i_p la_ = (lv4i_p)&sA[(buf)][ldRow * LDSW + ldHalf];               \
        lv4i_p lb_ = (lv4i_p)&sB[(buf)][ldRow * LDSW + ldHalf];               \
        __builtin_amdgcn_global_load_async_to_lds_b128(ga_, la_, 0, 0);       \
        __builtin_amdgcn_global_load_async_to_lds_b128(ga_, la_, 16, 0);      \
        __builtin_amdgcn_global_load_async_to_lds_b128(gb_, lb_, 0, 0);       \
        __builtin_amdgcn_global_load_async_to_lds_b128(gb_, lb_, 16, 0);      \
    } while (0)

    ISSUE_TILE(0, 0);                       // prologue: 4 async ops in flight

    for (int k0 = 0; k0 < N; k0 += BK) {
        const int cur = (k0 / BK) & 1;
        const int nxt = cur ^ 1;

        if (k0 + BK < N) {
            ISSUE_TILE(nxt, k0 + BK);       // run one K-tile ahead
            __builtin_amdgcn_s_wait_asynccnt(4);  // cur tile landed in LDS
        } else {
            __builtin_amdgcn_s_wait_asynccnt(0);
        }
        if (k0 + 2 * BK < N) {              // warm L2 two tiles ahead
            __builtin_prefetch(gA + k0 + 2 * BK, 0, 3);
            __builtin_prefetch(gB + k0 + 2 * BK, 0, 3);
        }
        __syncthreads();                    // cur tile visible to all waves

        Frag aF[2], bF[4];
#pragma unroll
        for (int i = 0; i < 2; ++i) {
            const __bf16* p = &sA[cur][(wm * 32 + i * 16 + lm) * LDSW];
            aF[i].u[0] = *(const uint4*)(p + lh * 16);
            aF[i].u[1] = *(const uint4*)(p + lh * 16 + 8);
        }
#pragma unroll
        for (int j = 0; j < 4; ++j) {
            const __bf16* p = &sB[cur][(wn * 64 + j * 16 + lm) * LDSW + lh * 16];
            bF[j].u[0] = ((const uint4*)p)[0];
            bF[j].u[1] = ((const uint4*)p)[1];
        }

#pragma unroll
        for (int i = 0; i < 2; ++i)
#pragma unroll
            for (int j = 0; j < 4; ++j)
                acc[i][j] = __builtin_amdgcn_wmma_f32_16x16x32_bf16(
                    false, aF[i].v, false, bF[j].v,
                    (short)0, acc[i][j], false, false);

        __syncthreads();                    // cur buffer free for reuse
    }
#undef ISSUE_TILE
#else
    __shared__ __bf16 sA[BM * LDSW];
    __shared__ __bf16 sB[BN * LDSW];

    for (int k0 = 0; k0 < N; k0 += BK) {
        const uint4* pa = (const uint4*)(gA + k0);
        const uint4* pb = (const uint4*)(gB + k0);
        const uint4 a0 = pa[0], a1 = pa[1];
        const uint4 b0 = pb[0], b1 = pb[1];

        if (k0 + BK < N) {
            __builtin_prefetch(gA + k0 + BK, 0, 3);
            __builtin_prefetch(gB + k0 + BK, 0, 3);
        }

        __syncthreads();
        *(uint4*)&sA[ldRow * LDSW + ldHalf]     = a0;
        *(uint4*)&sA[ldRow * LDSW + ldHalf + 8] = a1;
        *(uint4*)&sB[ldRow * LDSW + ldHalf]     = b0;
        *(uint4*)&sB[ldRow * LDSW + ldHalf + 8] = b1;
        __syncthreads();

        Frag aF[2], bF[4];
#pragma unroll
        for (int i = 0; i < 2; ++i) {
            const __bf16* p = &sA[(wm * 32 + i * 16 + lm) * LDSW];
            aF[i].u[0] = *(const uint4*)(p + lh * 16);
            aF[i].u[1] = *(const uint4*)(p + lh * 16 + 8);
        }
#pragma unroll
        for (int j = 0; j < 4; ++j) {
            const __bf16* p = &sB[(wn * 64 + j * 16 + lm) * LDSW + lh * 16];
            bF[j].u[0] = ((const uint4*)p)[0];
            bF[j].u[1] = ((const uint4*)p)[1];
        }

#pragma unroll
        for (int i = 0; i < 2; ++i)
#pragma unroll
            for (int j = 0; j < 4; ++j)
                acc[i][j] = __builtin_amdgcn_wmma_f32_16x16x32_bf16(
                    false, aF[i].v, false, bF[j].v,
                    (short)0, acc[i][j], false, false);
    }
#endif

    // C/D layout: VGPR r, lane l -> m = r + (l>>4)*8, n = l&15.
#pragma unroll
    for (int i = 0; i < 2; ++i) {
#pragma unroll
        for (int j = 0; j < 4; ++j) {
            const int zr = rowBase + wm * 32 + i * 16 + lh * 8;
            const int zc = colBase + wn * 64 + j * 16 + lm;
            float* zp = Z + (size_t)zr * Mout + zc;
#pragma unroll
            for (int r = 0; r < 8; ++r)
                zp[(size_t)r * Mout] = acc[i][j][r];
        }
    }
}

// ---------------------------------------------------------------------------
// Kernel 4: y = FWHT_M(z) * (1/sqrt(M)) * SU * Wscale, in place on d_out.
// One block per row; M = 16384 (64 KB f32 in LDS, fits 320 KB WGP LDS).
// ---------------------------------------------------------------------------
__global__ __launch_bounds__(1024) void fwht_z_kernel(
    float* __restrict__ Z, const float* __restrict__ SU,
    const float* __restrict__ wscale_p, int M)
{
    __shared__ float lds[16384];
    const int row = blockIdx.x;
    float* zr = Z + (size_t)row * M;

    for (int i = threadIdx.x; i < M; i += 1024)
        lds[i] = zr[i];
    __syncthreads();

    for (int h = 1; h < M; h <<= 1) {
        for (int t = threadIdx.x; t < (M >> 1); t += 1024) {
            const int i = ((t & ~(h - 1)) << 1) | (t & (h - 1));
            const float a = lds[i], b = lds[i + h];
            lds[i]     = a + b;
            lds[i + h] = a - b;
        }
        __syncthreads();
    }

    const float s = rsqrtf((float)M) * wscale_p[0];
    for (int i = threadIdx.x; i < M; i += 1024)
        zr[i] = lds[i] * s * SU[i];
}

// ---------------------------------------------------------------------------
extern "C" void kernel_launch(void* const* d_in, const int* in_sizes, int n_in,
                              void* d_out, int out_size, void* d_ws, size_t ws_size,
                              hipStream_t stream)
{
    const float* x      = (const float*)d_in[0];
    const int*   Qidxs  = (const int*)  d_in[1];
    const int*   Qidxs2 = (const int*)  d_in[2];
    const float* SU     = (const float*)d_in[3];
    const float* SV     = (const float*)d_in[4];
    const float* cb1    = (const float*)d_in[5];
    const float* cb2    = (const float*)d_in[6];
    const float* Wscale = (const float*)d_in[7];
    const float* irs    = (const float*)d_in[8];
    float* out = (float*)d_out;

    const int N    = in_sizes[4];            // 4096
    const int Mout = in_sizes[3];            // 16384
    const int T    = in_sizes[0] / N;        // 8192

    __bf16* t_ws = (__bf16*)d_ws;                  // T x N bf16 (64 MB)
    __bf16* w_ws = t_ws + (size_t)T * N;           // Mout x N bf16 (128 MB)

    // 1) input Hadamard transform -> bf16
    fwht_x_kernel<<<T, 256, 0, stream>>>(x, SV, t_ws, N);

    // 2) dequantize weights -> bf16
    const long long ngroups = (long long)Mout * N / 8;
    dequant_kernel<<<(int)((ngroups + 255) / 256), 256, 0, stream>>>(
        Qidxs, Qidxs2, cb1, cb2, irs, w_ws, ngroups);

    // 3) z = t @ W^T (WMMA bf16), written to d_out
    dim3 grid(Mout / BN, T / BM);
    gemm_kernel<<<grid, 256, 0, stream>>>(t_ws, w_ws, out, Mout, N);

    // 4) output Hadamard transform + scaling, in place on d_out
    fwht_z_kernel<<<T, 1024, 0, stream>>>(out, SU, Wscale, Mout);
}
